// GATPSPGNN_635655160274
// MI455X (gfx1250) — matrix operand order repaired and checked by reading, not compiled, not run on hardware
//
#include <hip/hip_runtime.h>
#include <math.h>

#define NNODES 50000
#define NEDGES 800000
#define NEG_SLOPE 0.2f

typedef __attribute__((ext_vector_type(16))) __bf16 v16bf;
typedef __attribute__((ext_vector_type(8)))  __bf16 v8bf;
typedef __attribute__((ext_vector_type(8)))  float  v8f;

// ---------- helpers ----------
static __device__ __forceinline__ unsigned short f32_to_bf16_rne(float f) {
    unsigned int u = __builtin_bit_cast(unsigned int, f);
    unsigned int r = u + 0x7fffu + ((u >> 16) & 1u);
    return (unsigned short)(r >> 16);
}
// monotone float<->uint mapping so unsigned atomicMax == float max
static __device__ __forceinline__ unsigned int flip_f32(float f) {
    unsigned int u = __builtin_bit_cast(unsigned int, f);
    return (u & 0x80000000u) ? ~u : (u | 0x80000000u);
}
static __device__ __forceinline__ float unflip_f32(unsigned int s) {
    unsigned int u = (s & 0x80000000u) ? (s ^ 0x80000000u) : ~s;
    return __builtin_bit_cast(float, u);
}
static __device__ __forceinline__ float leaky(float v) {
    return v > 0.f ? v : NEG_SLOPE * v;
}

// ---------- elementwise kernels ----------
__global__ void k_f32_to_bf16(const float* __restrict__ in, unsigned short* __restrict__ out, int n) {
    int i = blockIdx.x * blockDim.x + threadIdx.x;
    if (i < n) out[i] = f32_to_bf16_rne(in[i]);
}

__global__ void k_concat_bf16(const float* __restrict__ a, int ca,
                              const float* __restrict__ b, int cb,
                              unsigned short* __restrict__ out, int n) {
    int i = blockIdx.x * blockDim.x + threadIdx.x;
    int c = ca + cb;
    if (i < n * c) {
        int row = i / c, col = i - row * c;
        float v = (col < ca) ? a[(size_t)row * ca + col] : b[(size_t)row * cb + (col - ca)];
        out[i] = f32_to_bf16_rne(v);
    }
}

// repack W[K,NC] f32 -> fragment-major bf16: Wp[((k/32)*NC + col)*32 + (k%32)]
__global__ void k_pack_w_bf16(const float* __restrict__ W, unsigned short* __restrict__ Wp,
                              int K, int NC) {
    int i = blockIdx.x * blockDim.x + threadIdx.x;
    if (i < K * NC) {
        int k = i / NC, col = i - k * NC;
        Wp[((size_t)(k >> 5) * NC + col) * 32 + (k & 31)] = f32_to_bf16_rne(W[i]);
    }
}

__global__ void k_copy_f32(const float* __restrict__ in, float* __restrict__ out, int n) {
    int i = blockIdx.x * blockDim.x + threadIdx.x;
    if (i < n) out[i] = in[i];
}
__global__ void k_zero_u32(unsigned int* p, int n) {
    int i = blockIdx.x * blockDim.x + threadIdx.x;
    if (i < n) p[i] = 0u;           // flipped encoding of the float minimum
}
__global__ void k_zero_f32(float* p, int n) {
    int i = blockIdx.x * blockDim.x + threadIdx.x;
    if (i < n) p[i] = 0.f;
}
__global__ void k_elu(const float* __restrict__ in, float* __restrict__ out, int n) {
    int i = blockIdx.x * blockDim.x + threadIdx.x;
    if (i < n) { float x = in[i]; out[i] = x > 0.f ? x : (__expf(x) - 1.f); }
}
__global__ void k_tanh(const float* __restrict__ in, float* __restrict__ out, int n) {
    int i = blockIdx.x * blockDim.x + threadIdx.x;
    if (i < n) out[i] = tanhf(in[i]);
}

// ---------- bf16 WMMA GEMM (fully unrolled, packed B, optional dual weights) ----------
// C1[M,NC] = X[M,K] @ W1;  if DUAL also C2 = X @ W2 (A fragments reused).
// Block = 128 threads (4 waves); wave w covers cols {w*16 + t*64 : t<NC/64}; 16 rows/block.
template <int K, int NC, bool DUAL>
__global__ void __launch_bounds__(128)
k_gemm_wmma(const unsigned short* __restrict__ Xu,
            const unsigned short* __restrict__ Wp1u,
            const unsigned short* __restrict__ Wp2u,
            float* __restrict__ C1, float* __restrict__ C2) {
    constexpr int NT = NC / 64;     // 16-col tiles per wave
    constexpr int KS = K / 32;      // k steps
    const __bf16* X  = (const __bf16*)Xu;
    const __bf16* W1 = (const __bf16*)Wp1u;
    const __bf16* W2 = (const __bf16*)Wp2u;
    const int lane = threadIdx.x & 31;
    const int wave = threadIdx.x >> 5;
    const int l    = lane & 15;
    const int hf   = lane >> 4;
    const int m0   = blockIdx.x * 16;
    const __bf16* Xrow = X + (size_t)(m0 + l) * K;

    v8f acc1[NT], acc2[NT];
#pragma unroll
    for (int t = 0; t < NT; ++t) { acc1[t] = (v8f){}; acc2[t] = (v8f){}; }

#pragma unroll
    for (int ks = 0; ks < KS; ++ks) {
        // A 16x32 bf16 fragment: lanes0-15 K {k0..+7, k0+16..+23}; lanes16-31 +8
        const int aK = ks * 32 + hf * 8;
        v8bf alo = *(const v8bf*)(Xrow + aK);
        v8bf ahi = *(const v8bf*)(Xrow + aK + 16);
        v16bf A;
#pragma unroll
        for (int i = 0; i < 8; ++i) { A[i] = alo[i]; A[8 + i] = ahi[i]; }

#pragma unroll
        for (int t = 0; t < NT; ++t) {
            const int col = wave * 16 + t * 64 + l;
            const size_t base = ((size_t)ks * NC + col) * 32 + hf * 8;
            {
                v8bf blo = *(const v8bf*)(W1 + base);
                v8bf bhi = *(const v8bf*)(W1 + base + 16);
                v16bf B;
#pragma unroll
                for (int j = 0; j < 8; ++j) { B[j] = blo[j]; B[8 + j] = bhi[j]; }
                acc1[t] = __builtin_amdgcn_wmma_f32_16x16x32_bf16(
                    false, A, false, B, (short)0, acc1[t], false, false);
            }
            if (DUAL) {
                v8bf blo = *(const v8bf*)(W2 + base);
                v8bf bhi = *(const v8bf*)(W2 + base + 16);
                v16bf B;
#pragma unroll
                for (int j = 0; j < 8; ++j) { B[j] = blo[j]; B[8 + j] = bhi[j]; }
                acc2[t] = __builtin_amdgcn_wmma_f32_16x16x32_bf16(
                    false, A, false, B, (short)0, acc2[t], false, false);
            }
        }
    }
    // D layout: VGPR r -> row m0+r (+8 upper half-wave), col = tile + lane%16
#pragma unroll
    for (int t = 0; t < NT; ++t) {
        const int col = wave * 16 + t * 64 + l;
#pragma unroll
        for (int r = 0; r < 8; ++r) {
            const size_t row = (size_t)(m0 + r + hf * 8);
            C1[row * NC + col] = acc1[t][r];
            if (DUAL) C2[row * NC + col] = acc2[t][r];
        }
    }
}

// ---------- attention ----------
__global__ void k_elr(const float* __restrict__ h, const float* __restrict__ al,
                      const float* __restrict__ ar, float* __restrict__ el,
                      float* __restrict__ er, int n, int H, int D) {
    int i = blockIdx.x * blockDim.x + threadIdx.x;
    if (i < n * H) {
        int node = i / H, hh = i - node * H;
        const float* hp  = h + (size_t)node * H * D + (size_t)hh * D;
        const float* alp = al + hh * D;
        const float* arp = ar + hh * D;
        float sl = 0.f, sr = 0.f;
        for (int d = 0; d < D; ++d) { float v = hp[d]; sl += v * alp[d]; sr += v * arp[d]; }
        el[i] = sl; er[i] = sr;
    }
}

__global__ void k_edge_max(const int* __restrict__ src, const int* __restrict__ dst,
                           const float* __restrict__ el, const float* __restrict__ er,
                           unsigned int* __restrict__ emax, int E, int H) {
    int i = blockIdx.x * blockDim.x + threadIdx.x;
    if (i < E * H) {
        int e = i / H, hh = i - e * H;
        int s = src[e], d = dst[e];
        float v = leaky(el[s * H + hh] + er[d * H + hh]);
        atomicMax(&emax[d * H + hh], flip_f32(v));
    }
}

__global__ void k_edge_expsum(const int* __restrict__ src, const int* __restrict__ dst,
                              const float* __restrict__ el, const float* __restrict__ er,
                              const unsigned int* __restrict__ emax,
                              float* __restrict__ ee, float* __restrict__ denom, int E, int H) {
    int i = blockIdx.x * blockDim.x + threadIdx.x;
    if (i < E * H) {
        int e = i / H, hh = i - e * H;
        int s = src[e], d = dst[e];
        float v = leaky(el[s * H + hh] + er[d * H + hh]);
        float t = __expf(v - unflip_f32(emax[d * H + hh]));
        ee[i] = t;
        atomicAdd(&denom[d * H + hh], t);
    }
}

// out[dst] += alpha * h[src], float4 gathers, 4 atomics per thread
__global__ void k_edge_scatter(const int* __restrict__ src, const int* __restrict__ dst,
                               const float* __restrict__ hbuf, const float* __restrict__ ee,
                               const float* __restrict__ denom, float* __restrict__ out,
                               int E, int H, int D) {
    const int C = H * D;
    const int nq = C >> 2;
    long i = (long)blockIdx.x * blockDim.x + threadIdx.x;
    if (i < (long)E * nq) {
        int e = (int)(i / nq);
        int q = (int)(i - (long)e * nq);
        int j = q << 2;
        int hh = j / D;
        int s = src[e], d = dst[e];
        float alpha = ee[e * H + hh] / denom[d * H + hh];
        const float4 hv = *(const float4*)(hbuf + (size_t)s * C + j);
        float* op = out + (size_t)d * C + j;
        atomicAdd(op + 0, hv.x * alpha);
        atomicAdd(op + 1, hv.y * alpha);
        atomicAdd(op + 2, hv.z * alpha);
        atomicAdd(op + 3, hv.w * alpha);
    }
}

// final: d_out[0:N*64) = mean over heads of elu(gat2 out); d_out[N*64:2*N*64) = h_p
__global__ void k_final(const float* __restrict__ o, const float* __restrict__ hp,
                        float* __restrict__ dout, int n) {
    int i = blockIdx.x * blockDim.x + threadIdx.x;
    int halfn = n * 64;
    if (i < halfn) {
        int node = i >> 6, d = i & 63;
        float a = o[(size_t)node * 128 + d];
        float b = o[(size_t)node * 128 + 64 + d];
        a = a > 0.f ? a : (__expf(a) - 1.f);
        b = b > 0.f ? b : (__expf(b) - 1.f);
        dout[i] = 0.5f * (a + b);
    } else if (i < 2 * halfn) {
        dout[i] = hp[i - halfn];
    }
}

// ---------- host orchestration ----------
extern "C" void kernel_launch(void* const* d_in, const int* in_sizes, int n_in,
                              void* d_out, int out_size, void* d_ws, size_t ws_size,
                              hipStream_t stream) {
    (void)in_sizes; (void)n_in; (void)out_size; (void)ws_size;
    const float* fvs = (const float*)d_in[0];
    const float* pos = (const float*)d_in[1];
    const int*   src = (const int*)d_in[2];
    const int*   dst = (const int*)d_in[3];
    const float* gW[3]  = {(const float*)d_in[4],  (const float*)d_in[8],  (const float*)d_in[12]};
    const float* gal[3] = {(const float*)d_in[5],  (const float*)d_in[9],  (const float*)d_in[13]};
    const float* gar[3] = {(const float*)d_in[6],  (const float*)d_in[10], (const float*)d_in[14]};
    const float* grW[3] = {(const float*)d_in[7],  (const float*)d_in[11], (const float*)d_in[15]};
    const float* pW[2]  = {(const float*)d_in[16], (const float*)d_in[19]};
    const float* pal[2] = {(const float*)d_in[17], (const float*)d_in[20]};
    const float* par[2] = {(const float*)d_in[18], (const float*)d_in[21]};

    const int N = NNODES, E = NEDGES;
    const int TB = 256;

    char* ws = (char*)d_ws; size_t off = 0;
    auto carve = [&](size_t bytes) -> char* {
        char* p = ws + off; off += (bytes + 255) & ~(size_t)255; return p;
    };
    unsigned short* xcat = (unsigned short*)carve((size_t)N * 192 * 2);
    unsigned short* xpg  = (unsigned short*)carve((size_t)N * 64 * 2);
    unsigned short* wbf  = (unsigned short*)carve((size_t)192 * 128 * 2);
    unsigned short* rwbf = (unsigned short*)carve((size_t)192 * 128 * 2);
    float* hbuf   = (float*)carve((size_t)N * 128 * 4);
    float* outbuf = (float*)carve((size_t)N * 128 * 4);
    float* el     = (float*)carve((size_t)N * 2 * 4);
    float* er     = (float*)carve((size_t)N * 2 * 4);
    unsigned int* emax = (unsigned int*)carve((size_t)N * 2 * 4);
    float* denom  = (float*)carve((size_t)N * 2 * 4);
    float* ee     = (float*)carve((size_t)E * 2 * 4);
    float* hs     = (float*)carve((size_t)N * 128 * 4);
    float* hpa    = (float*)carve((size_t)N * 64 * 4);
    float* hpb    = (float*)carve((size_t)N * 64 * 4);

    auto cdiv = [](long a, long b) -> int { return (int)((a + b - 1) / b); };

    // rWf != nullptr  -> GAT shape (K=192, C=128, dual GEMM: h + residual)
    // rWf == nullptr  -> PGNN shape (K=64, C=64, identity residual)
    auto run_conv = [&](const unsigned short* xbf, int K, const float* Wf, const float* rWf,
                        const float* identRes, const float* alf, const float* arf,
                        int H, int D) {
        const int C = H * D;
        k_pack_w_bf16<<<cdiv((long)K * C, TB), TB, 0, stream>>>(Wf, wbf, K, C);
        if (rWf) {
            k_pack_w_bf16<<<cdiv((long)K * C, TB), TB, 0, stream>>>(rWf, rwbf, K, C);
            k_gemm_wmma<192, 128, true><<<dim3(N / 16), 128, 0, stream>>>(
                xbf, wbf, rwbf, hbuf, outbuf);
        } else {
            k_copy_f32<<<cdiv((long)N * C, TB), TB, 0, stream>>>(identRes, outbuf, N * C);
            k_gemm_wmma<64, 64, false><<<dim3(N / 16), 128, 0, stream>>>(
                xbf, wbf, nullptr, hbuf, nullptr);
        }
        k_elr<<<cdiv((long)N * H, TB), TB, 0, stream>>>(hbuf, alf, arf, el, er, N, H, D);
        k_zero_u32<<<cdiv((long)N * H, TB), TB, 0, stream>>>(emax, N * H);
        k_zero_f32<<<cdiv((long)N * H, TB), TB, 0, stream>>>(denom, N * H);
        k_edge_max<<<cdiv((long)E * H, TB), TB, 0, stream>>>(src, dst, el, er, emax, E, H);
        k_edge_expsum<<<cdiv((long)E * H, TB), TB, 0, stream>>>(src, dst, el, er, emax, ee, denom, E, H);
        k_edge_scatter<<<cdiv((long)E * (C / 4), TB), TB, 0, stream>>>(src, dst, hbuf, ee, denom, outbuf, E, H, D);
    };

    // ---- layer 0 ----
    k_concat_bf16<<<cdiv((long)N * 192, TB), TB, 0, stream>>>(fvs, 128, pos, 64, xcat, N);
    run_conv(xcat, 192, gW[0], grW[0], nullptr, gal[0], gar[0], 2, 64);
    k_elu<<<cdiv((long)N * 128, TB), TB, 0, stream>>>(outbuf, hs, N * 128);

    k_f32_to_bf16<<<cdiv((long)N * 64, TB), TB, 0, stream>>>(pos, xpg, N * 64);
    run_conv(xpg, 64, pW[0], nullptr, pos, pal[0], par[0], 2, 32);
    k_tanh<<<cdiv((long)N * 64, TB), TB, 0, stream>>>(outbuf, hpa, N * 64);

    // ---- layer 1 ----
    k_concat_bf16<<<cdiv((long)N * 192, TB), TB, 0, stream>>>(hs, 128, hpa, 64, xcat, N);
    run_conv(xcat, 192, gW[1], grW[1], nullptr, gal[1], gar[1], 2, 64);
    k_elu<<<cdiv((long)N * 128, TB), TB, 0, stream>>>(outbuf, hs, N * 128);

    k_f32_to_bf16<<<cdiv((long)N * 64, TB), TB, 0, stream>>>(hpa, xpg, N * 64);
    run_conv(xpg, 64, pW[1], nullptr, hpa, pal[1], par[1], 2, 32);
    k_tanh<<<cdiv((long)N * 64, TB), TB, 0, stream>>>(outbuf, hpb, N * 64);

    // ---- output layer ----
    k_concat_bf16<<<cdiv((long)N * 192, TB), TB, 0, stream>>>(hs, 128, hpb, 64, xcat, N);
    run_conv(xcat, 192, gW[2], grW[2], nullptr, gal[2], gar[2], 2, 64);
    k_final<<<cdiv((long)N * 128, TB), TB, 0, stream>>>(outbuf, hpb, (float*)d_out, N);
}
